// Classifiers_18133351923786
// MI455X (gfx1250) — matrix operand order repaired and checked by reading, not compiled, not run on hardware
//
#include <hip/hip_runtime.h>
#include <hip/hip_bf16.h>

typedef __bf16 bf16_t;
typedef __attribute__((ext_vector_type(16))) __bf16 v16bf;
typedef __attribute__((ext_vector_type(8)))  __bf16 v8bf;
typedef __attribute__((ext_vector_type(8)))  float  v8f;

__device__ __forceinline__ v16bf join16(v8bf lo, v8bf hi) {
    union { v16bf v; v8bf h[2]; } u;
    u.h[0] = lo; u.h[1] = hi;
    return u.v;
}

// A fragment: 16(M) x 32(K) bf16 tile from row-major [row][k] with row stride `stride` elems.
// ISA 16-bit A layout: lanes 0-15 (M=lane) hold K=kb+0..7 (v0-3) and K=kb+16..23 (v4-7);
// lanes 16-31 hold K=kb+8..15 and K=kb+24..31.
__device__ __forceinline__ v16bf load_a_frag(const bf16_t* base, int stride, int kbase, int lane) {
    int m  = lane & 15;
    int kh = (lane >> 4) << 3;            // 0 or 8
    const bf16_t* p = base + m * stride + kbase + kh;
    v8bf lo = *(const v8bf*)(p);          // K = kb+kh+0..7
    v8bf hi = *(const v8bf*)(p + 16);     // K = kb+kh+16..23
    return join16(lo, hi);
}

// B fragment: 32(K) x 16(N) tile, weights stored TRANSPOSED as Wt[n][k], row stride `stride`.
// B layout (matches A): lanes 0-15 (N=lane) hold K=kb+0..7 and kb+16..23; lanes 16-31 hold K=kb+8..15 and kb+24..31.
__device__ __forceinline__ v16bf load_b_frag(const bf16_t* base, int stride, int kbase, int lane) {
    int n  = lane & 15;
    int kh = (lane >> 4) << 3;            // 0 or 8
    const bf16_t* p = base + n * stride + kbase + kh;
    v8bf lo = *(const v8bf*)(p);
    v8bf hi = *(const v8bf*)(p + 16);
    return join16(lo, hi);
}

__device__ __forceinline__ v8f wmma_bf16(v16bf a, v16bf b, v8f c) {
    return __builtin_amdgcn_wmma_f32_16x16x32_bf16(false, a, false, b, (short)0, c, false, false);
}

__device__ __forceinline__ float sigmoidf_(float x) { return 1.0f / (1.0f + __expf(-x)); }

// Pack f32 weight [K][N] (row-major) -> bf16 transposed padded Wt[NPAD][KPAD].
__global__ void pack_weight_kernel(const float* __restrict__ src, bf16_t* __restrict__ dst,
                                   int K, int N, int KPAD, int NPAD) {
    int idx = blockIdx.x * blockDim.x + threadIdx.x;
    if (idx >= KPAD * NPAD) return;
    int nrow = idx / KPAD, kcol = idx - nrow * KPAD;
    float v = (nrow < N && kcol < K) ? src[(size_t)kcol * N + nrow] : 0.0f;
    dst[idx] = (bf16_t)v;
}

// Keras-style relu-LSTM (gate order i,f,g,o). One wave = 16 batch rows.
// x: [B,T,KIN] (TIN = float or bf16). Wt: [4U][KPAD], Ut: [4U][U] packed bf16. bias: [4U] f32.
template<typename TIN, int T, int KIN, int KPAD, int U, bool RETSEQ>
__global__ __launch_bounds__(128) void lstm_kernel(
    const TIN* __restrict__ x, const bf16_t* __restrict__ Wt, const bf16_t* __restrict__ Ut,
    const float* __restrict__ bias, bf16_t* __restrict__ hout) {
    constexpr int NW  = 4;
    constexpr int U16 = U / 16;
    __shared__ alignas(16) bf16_t xs[NW][16 * KPAD];
    __shared__ alignas(16) bf16_t hs[NW][2][16 * U];
    const int wave = threadIdx.x >> 5;
    const int lane = threadIdx.x & 31;
    const long b0  = ((long)blockIdx.x * NW + wave) * 16;
    const int n  = lane & 15;
    const int hi = lane >> 4;

    float c[U16][8];
    #pragma unroll
    for (int j = 0; j < U16; ++j)
        #pragma unroll
        for (int r = 0; r < 8; ++r) c[j][r] = 0.0f;

    for (int t = 0; t < T; ++t) {
        // Stage x_t into LDS as bf16, zero-padded KIN..KPAD.
        for (int idx = lane; idx < 16 * KPAD; idx += 32) {
            int row = idx / KPAD, col = idx - row * KPAD;
            float v = 0.0f;
            if (col < KIN) v = (float)x[(((b0 + row) * T) + t) * KIN + col];
            xs[wave][idx] = (bf16_t)v;
        }
        __syncthreads();

        const int wb = t & 1;       // h_t write buffer
        const int rb = wb ^ 1;      // h_{t-1} read buffer

        #pragma unroll
        for (int jt = 0; jt < U16; ++jt) {
            v8f acc[4];
            #pragma unroll
            for (int g = 0; g < 4; ++g)
                #pragma unroll
                for (int r = 0; r < 8; ++r) acc[g][r] = 0.0f;

            #pragma unroll
            for (int k = 0; k < KPAD / 32; ++k) {
                v16bf a = load_a_frag(&xs[wave][0], KPAD, k * 32, lane);
                #pragma unroll
                for (int g = 0; g < 4; ++g) {
                    v16bf b = load_b_frag(Wt + (size_t)(g * U + jt * 16) * KPAD, KPAD, k * 32, lane);
                    acc[g] = wmma_bf16(a, b, acc[g]);
                }
            }
            if (t > 0) {
                #pragma unroll
                for (int k = 0; k < U / 32; ++k) {
                    v16bf a = load_a_frag(&hs[wave][rb][0], U, k * 32, lane);
                    #pragma unroll
                    for (int g = 0; g < 4; ++g) {
                        v16bf b = load_b_frag(Ut + (size_t)(g * U + jt * 16) * U, U, k * 32, lane);
                        acc[g] = wmma_bf16(a, b, acc[g]);
                    }
                }
            }
            const int col = jt * 16 + n;
            const float bi  = bias[0 * U + col];
            const float bff = bias[1 * U + col];
            const float bg  = bias[2 * U + col];
            const float bo  = bias[3 * U + col];
            #pragma unroll
            for (int r = 0; r < 8; ++r) {
                float ig = sigmoidf_(acc[0][r] + bi);
                float fg = sigmoidf_(acc[1][r] + bff);
                float gg = fmaxf(acc[2][r] + bg, 0.0f);
                float og = sigmoidf_(acc[3][r] + bo);
                float cn = fg * c[jt][r] + ig * gg;
                c[jt][r] = cn;
                float h = fmaxf(cn, 0.0f) * og;
                int m = r + hi * 8;
                hs[wave][wb][m * U + col] = (bf16_t)h;
                if (RETSEQ) {
                    hout[(((b0 + m) * T) + t) * U + col] = (bf16_t)h;
                } else if (t == T - 1) {
                    hout[(b0 + m) * U + col] = (bf16_t)h;
                }
            }
        }
        __syncthreads();
    }
}

// out[B,NPAD] = relu(A[B,K] @ Wd + b), Wt packed [NPAD][K]; padded cols come out 0.
template<int K, int NPAD, int NREAL>
__global__ __launch_bounds__(128) void dense_relu_kernel(
    const bf16_t* __restrict__ A, const bf16_t* __restrict__ Wt,
    const float* __restrict__ bias, bf16_t* __restrict__ out) {
    const int wave = threadIdx.x >> 5, lane = threadIdx.x & 31;
    const long b0 = ((long)blockIdx.x * 4 + wave) * 16;
    const int n = lane & 15, hi = lane >> 4;
    const bf16_t* arow = A + (size_t)b0 * K;
    #pragma unroll
    for (int nt = 0; nt < NPAD / 16; ++nt) {
        v8f acc;
        #pragma unroll
        for (int r = 0; r < 8; ++r) acc[r] = 0.0f;
        #pragma unroll
        for (int k = 0; k < K / 32; ++k) {
            v16bf a = load_a_frag(arow, K, k * 32, lane);
            v16bf b = load_b_frag(Wt + (size_t)nt * 16 * K, K, k * 32, lane);
            acc = wmma_bf16(a, b, acc);
        }
        int col = nt * 16 + n;
        float bv = (col < NREAL) ? bias[col] : 0.0f;
        #pragma unroll
        for (int r = 0; r < 8; ++r) {
            int m = r + hi * 8;
            out[(b0 + m) * NPAD + col] = (bf16_t)fmaxf(acc[r] + bv, 0.0f);
        }
    }
}

// out[B,NOUT] = softmax(A[B,K] @ Wd + b); single 16-col N tile (NOUT <= 16).
// Row-wise reduction uses shfl_xor within each 16-lane half (D layout: row spread over 16 lanes).
template<int K, int NOUT>
__global__ __launch_bounds__(128) void dense_softmax_kernel(
    const bf16_t* __restrict__ A, const bf16_t* __restrict__ Wt,
    const float* __restrict__ bias, float* __restrict__ out) {
    const int wave = threadIdx.x >> 5, lane = threadIdx.x & 31;
    const long b0 = ((long)blockIdx.x * 4 + wave) * 16;
    const int n = lane & 15, hi = lane >> 4;
    const bf16_t* arow = A + (size_t)b0 * K;
    v8f acc;
    #pragma unroll
    for (int r = 0; r < 8; ++r) acc[r] = 0.0f;
    #pragma unroll
    for (int k = 0; k < K / 32; ++k) {
        v16bf a = load_a_frag(arow, K, k * 32, lane);
        v16bf b = load_b_frag(Wt, K, k * 32, lane);
        acc = wmma_bf16(a, b, acc);
    }
    float bv = (n < NOUT) ? bias[n] : 0.0f;
    #pragma unroll
    for (int r = 0; r < 8; ++r) {
        float v = (n < NOUT) ? (acc[r] + bv) : -1.0e30f;
        float mx = v;
        #pragma unroll
        for (int mk = 1; mk < 16; mk <<= 1) mx = fmaxf(mx, __shfl_xor(mx, mk, 32));
        float e = __expf(v - mx);        // padded lanes: exp(-1e30 - mx) -> 0
        float s = e;
        #pragma unroll
        for (int mk = 1; mk < 16; mk <<= 1) s += __shfl_xor(s, mk, 32);
        if (n < NOUT) out[(b0 + r + hi * 8) * (size_t)NOUT + n] = e / s;
    }
}

extern "C" void kernel_launch(void* const* d_in, const int* in_sizes, int n_in,
                              void* d_out, int out_size, void* d_ws, size_t ws_size,
                              hipStream_t stream) {
    (void)n_in; (void)out_size; (void)ws_size;
    const float* input_ac = (const float*)d_in[0];
    const float* input_cc = (const float*)d_in[1];
    const float* W1  = (const float*)d_in[2];
    const float* U1  = (const float*)d_in[3];
    const float* b1  = (const float*)d_in[4];
    const float* W2  = (const float*)d_in[5];
    const float* U2  = (const float*)d_in[6];
    const float* b2  = (const float*)d_in[7];
    const float* W3  = (const float*)d_in[8];
    const float* U3  = (const float*)d_in[9];
    const float* b3  = (const float*)d_in[10];
    const float* Wd1 = (const float*)d_in[11];
    const float* bd1 = (const float*)d_in[12];
    const float* Wd2 = (const float*)d_in[13];
    const float* bd2 = (const float*)d_in[14];
    const float* Wcc = (const float*)d_in[15];
    const float* Ucc = (const float*)d_in[16];
    const float* bcc = (const float*)d_in[17];
    const float* Wdcc= (const float*)d_in[18];
    const float* bdcc= (const float*)d_in[19];

    const int Bsz = in_sizes[0] / (3 * 132);   // 65536
    char* ws = (char*)d_ws;
    size_t off = 0;
    auto take = [&](size_t bytes) { size_t r = off; off += (bytes + 255) & ~(size_t)255; return r; };

    // Packed bf16 transposed weights
    size_t oW1t  = take(256 * 160 * 2), oU1t  = take(256 * 64 * 2);
    size_t oW2t  = take(512 * 64 * 2),  oU2t  = take(512 * 128 * 2);
    size_t oW3t  = take(256 * 128 * 2), oU3t  = take(256 * 64 * 2);
    size_t oWd1t = take(128 * 64 * 2),  oWd2t = take(16 * 128 * 2);
    size_t oWcct = take(256 * 160 * 2), oUcct = take(256 * 64 * 2);
    size_t oWdcct= take(16 * 64 * 2);
    // Activation buffers (stream-ordered reuse: h3 over h1seq, z over h2seq)
    size_t oH1  = take((size_t)Bsz * 3 * 64  * 2);
    size_t oH2  = take((size_t)Bsz * 3 * 128 * 2);
    size_t oHcc = take((size_t)Bsz * 64 * 2);
    size_t oH3  = oH1;   // h1seq dead after lstm2
    size_t oZ   = oH2;   // h2seq dead after lstm3

    auto bfp = [&](size_t o) { return (bf16_t*)(ws + o); };
    auto pack = [&](const float* src, size_t dsto, int K, int N, int KPAD, int NPAD) {
        int total = KPAD * NPAD;
        pack_weight_kernel<<<(total + 255) / 256, 256, 0, stream>>>(src, bfp(dsto), K, N, KPAD, NPAD);
    };
    pack(W1,  oW1t,  132, 256, 160, 256);
    pack(U1,  oU1t,  64,  256, 64,  256);
    pack(W2,  oW2t,  64,  512, 64,  512);
    pack(U2,  oU2t,  128, 512, 128, 512);
    pack(W3,  oW3t,  128, 256, 128, 256);
    pack(U3,  oU3t,  64,  256, 64,  256);
    pack(Wd1, oWd1t, 64,  100, 64,  128);
    pack(Wd2, oWd2t, 100, 11,  128, 16);
    pack(Wcc, oWcct, 132, 256, 160, 256);
    pack(Ucc, oUcct, 64,  256, 64,  256);
    pack(Wdcc,oWdcct,64,  10,  64,  16);

    const int blocks = Bsz / 64;   // 4 waves x 16 rows per block

    // ac branch
    lstm_kernel<float, 3, 132, 160, 64, true><<<blocks, 128, 0, stream>>>(
        input_ac, bfp(oW1t), bfp(oU1t), b1, bfp(oH1));
    lstm_kernel<bf16_t, 3, 64, 64, 128, true><<<blocks, 128, 0, stream>>>(
        bfp(oH1), bfp(oW2t), bfp(oU2t), b2, bfp(oH2));
    lstm_kernel<bf16_t, 3, 128, 128, 64, false><<<blocks, 128, 0, stream>>>(
        bfp(oH2), bfp(oW3t), bfp(oU3t), b3, bfp(oH3));
    dense_relu_kernel<64, 128, 100><<<blocks, 128, 0, stream>>>(
        bfp(oH3), bfp(oWd1t), bd1, bfp(oZ));
    float* out_ac = (float*)d_out;
    float* out_cc = out_ac + (size_t)Bsz * 11;
    dense_softmax_kernel<128, 11><<<blocks, 128, 0, stream>>>(
        bfp(oZ), bfp(oWd2t), bd2, out_ac);

    // cc branch
    lstm_kernel<float, 5, 132, 160, 64, false><<<blocks, 128, 0, stream>>>(
        input_cc, bfp(oWcct), bfp(oUcct), bcc, bfp(oHcc));
    dense_softmax_kernel<64, 10><<<blocks, 128, 0, stream>>>(
        bfp(oHcc), bfp(oWdcct), bdcc, out_cc);
}
